// Transformer_2946347565897
// MI455X (gfx1250) — matrix operand order repaired and checked by reading, not compiled, not run on hardware
//
#include <hip/hip_runtime.h>
#include <math.h>

typedef __bf16 bf16_t;
typedef __attribute__((ext_vector_type(16))) __bf16 v16bf;
typedef __attribute__((ext_vector_type(8)))  __bf16 v8bf;
typedef __attribute__((ext_vector_type(8)))  float  v8f;

#define WMMA_BF16(a, b, c) \
    __builtin_amdgcn_wmma_f32_16x16x32_bf16(false, (a), false, (b), (short)0, (c), false, false)

__device__ __forceinline__ v8f vzero8() {
    v8f v;
#pragma unroll
    for (int i = 0; i < 8; ++i) v[i] = 0.0f;
    return v;
}

// A-layout fragment (bf16 16x32): lane row = L&15; lanes 0-15 take K=0..15,
// lanes 16-31 take K=16..31. rowp points at (row, k=0). Two 16B loads.
__device__ __forceinline__ v16bf ldfragA(const bf16_t* rowp, int lane) {
    const bf16_t* p = rowp + ((lane & 16) ? 16 : 0);
    v8bf lo = *(const v8bf*)(p);
    v8bf hi = *(const v8bf*)(p + 8);
    return __builtin_shufflevector(lo, hi, 0, 1, 2, 3, 4, 5, 6, 7, 8, 9, 10, 11, 12, 13, 14, 15);
}

// B-layout fragment (bf16 32x16): lane col = L&15, K = 16 consecutive values
// starting at 0 (lanes 0-15) or 16 (lanes 16-31). p points at this lane's first element.
__device__ __forceinline__ v16bf ldfragB(const bf16_t* p) {
    v8bf lo = *(const v8bf*)(p);
    v8bf hi = *(const v8bf*)(p + 8);
    return __builtin_shufflevector(lo, hi, 0, 1, 2, 3, 4, 5, 6, 7, 8, 9, 10, 11, 12, 13, 14, 15);
}

// ---------------------------------------------------------------------------
// Weight convert + transpose: src[K,N] f32 -> dst[N,K] bf16
// ---------------------------------------------------------------------------
__global__ void cvt_transpose_kernel(const float* __restrict__ src, bf16_t* __restrict__ dst,
                                     int K, int N) {
    int idx = blockIdx.x * 256 + threadIdx.x;
    if (idx < K * N) {
        int k = idx / N, n = idx % N;
        dst[(size_t)n * K + k] = (bf16_t)src[idx];
    }
}

// ---------------------------------------------------------------------------
// Embedding gather + positional-imag (sin) generation
// ---------------------------------------------------------------------------
__global__ __launch_bounds__(256) void embed_pos_kernel(
    const int* __restrict__ tokens, const float* __restrict__ embed,
    float* __restrict__ zr_f, bf16_t* __restrict__ zr_b, bf16_t* __restrict__ zi_b,
    int S, int D) {
    int row = blockIdx.x;
    int s   = row % S;
    int tok = tokens[row];
    for (int d = threadIdx.x; d < D; d += 256) {
        float v = embed[(size_t)tok * D + d];
        zr_f[(size_t)row * D + d] = v;
        zr_b[(size_t)row * D + d] = (bf16_t)v;
        int p = d >> 1;
        float freq = __powf(10000.0f, -(2.0f * (float)p) / (float)D);
        zi_b[(size_t)row * D + d] = (bf16_t)__sinf((float)s * freq);  // GAMMA = 1
    }
}

// ---------------------------------------------------------------------------
// bf16 WMMA GEMM, B pre-transposed: Cout = (accum?Cin:0) + sign*A@B [+bias][relu]
// A[M,K] row-major, Bt[N,K] row-major. Block tile 128Mx64N, 4 waves,
// each wave 32Mx64N (8 WMMAs / 32-K step). A rows are wave-private ->
// fragments load straight from global; only the shared B tile goes through
// double-buffered LDS (single barrier per K-step: writes to buffer (i+1)&1
// after barrier i are race-free since its readers finished in iteration i-1).
// ---------------------------------------------------------------------------
#define GM 128
#define GN 64
#define GK 32

__global__ __launch_bounds__(128) void gemm_bf16_kernel(
    const bf16_t* __restrict__ A, const bf16_t* __restrict__ Bt,
    const float* __restrict__ Cin, float* __restrict__ Cout,
    bf16_t* __restrict__ Obf, const float* __restrict__ bias,
    int M, int N, int K, float sign, int accum, int relu) {
    __shared__ __align__(16) bf16_t Bs[2][GN * GK];

    const int bm = blockIdx.y * GM;
    const int bn = blockIdx.x * GN;
    const int tid  = threadIdx.x;
    const int wave = tid >> 5;
    const int lane = tid & 31;
    const int koff = (lane & 16) ? 16 : 0;

    // Wave-private A fragment row pointers (row = lane&15 within each 16-strip).
    const bf16_t* Ar0 = A + (size_t)(bm + wave * 32 + (lane & 15)) * K;
    const bf16_t* Ar1 = Ar0 + (size_t)16 * K;

    // Shared B staging: one 32B half-row of Bt per thread.
    const int brow = tid >> 1, bcb = (tid & 1) * 16;
    const bf16_t* Brow = Bt + (size_t)(bn + brow) * K + bcb;

    v8f acc[2][4];
#pragma unroll
    for (int s = 0; s < 2; ++s)
#pragma unroll
        for (int g = 0; g < 4; ++g) acc[s][g] = vzero8();

    uint4 breg[2];
    // prologue: B tile 0 -> regs -> LDS buffer 0
    {
        const uint4* sb = (const uint4*)(Brow);
        breg[0] = sb[0]; breg[1] = sb[1];
        uint4* db = (uint4*)(Bs[0] + brow * GK + bcb);
        db[0] = breg[0]; db[1] = breg[1];
    }

    const int nsteps = K / GK;
    for (int i = 0; i < nsteps; ++i) {
        __syncthreads();                       // buffer i&1 stores visible; prior reads done
        const int buf = i & 1;
        // Next B tile's global loads in flight under the WMMAs.
        if (i + 1 < nsteps) {
            const uint4* sb = (const uint4*)(Brow + (i + 1) * GK);
            breg[0] = sb[0]; breg[1] = sb[1];
        }
        if (i + 2 < nsteps) {
            __builtin_prefetch(Ar0 + (i + 2) * GK, 0, 1);
            __builtin_prefetch(Ar1 + (i + 2) * GK, 0, 1);
            __builtin_prefetch(Brow + (i + 2) * GK, 0, 1);
        }

        // A fragments: direct from global (wave-private rows).
        v16bf af[2];
        af[0] = ldfragA(Ar0 + i * GK, lane);
        af[1] = ldfragA(Ar1 + i * GK, lane);
#pragma unroll
        for (int g = 0; g < 4; ++g) {
            v16bf bfrag = ldfragB(&Bs[buf][(g * 16 + (lane & 15)) * GK + koff]);
#pragma unroll
            for (int s = 0; s < 2; ++s) acc[s][g] = WMMA_BF16(af[s], bfrag, acc[s][g]);
        }

        // Safe without a second barrier (see header comment).
        if (i + 1 < nsteps) {
            uint4* db = (uint4*)(Bs[(i + 1) & 1] + brow * GK + bcb);
            db[0] = breg[0]; db[1] = breg[1];
        }
    }

    // Epilogue: bias hoisted, Cin batched 8-wide so loads clause + single wait.
    const int half = (lane >> 4) & 1;
    const int ncol = bn + (lane & 15);
    float bvals[4];
#pragma unroll
    for (int g = 0; g < 4; ++g) bvals[g] = bias ? bias[ncol + g * 16] : 0.0f;
#pragma unroll
    for (int s = 0; s < 2; ++s) {
        const int mbase = bm + wave * 32 + s * 16 + half * 8;
#pragma unroll
        for (int g = 0; g < 4; ++g) {
            const int n = ncol + g * 16;
            float cin[8];
            if (accum) {
#pragma unroll
                for (int r = 0; r < 8; ++r) cin[r] = Cin[(size_t)(mbase + r) * N + n];
            }
#pragma unroll
            for (int r = 0; r < 8; ++r) {
                float v = sign * acc[s][g][r] + bvals[g];
                if (accum) v += cin[r];
                if (relu)  v = fmaxf(v, 0.0f);
                Cout[(size_t)(mbase + r) * N + n] = v;
                if (Obf) Obf[(size_t)(mbase + r) * N + n] = (bf16_t)v;
            }
        }
    }
}

// ---------------------------------------------------------------------------
// V transpose: v[b,s,h*DK+dk] -> vt[((b*H+h)*DK+dk)*S + s]
// ---------------------------------------------------------------------------
__global__ void transpose_v_kernel(const bf16_t* __restrict__ v, bf16_t* __restrict__ vt,
                                   int S, int D, int H, int DK) {
    int idx = blockIdx.x * 256 + threadIdx.x;
    int row = idx / D;       // b*S + s
    int col = idx % D;       // h*DK + dk
    int b = row / S, s = row % S;
    int h = col / DK, dk = col % DK;
    vt[(((size_t)b * H + h) * DK + dk) * S + s] = v[idx];
}

// ---------------------------------------------------------------------------
// Fused flash-style complex attention (wave32 per 16-query tile).
// ---------------------------------------------------------------------------
__global__ __launch_bounds__(32) void attn_kernel(
    const bf16_t* __restrict__ qr, const bf16_t* __restrict__ qi,
    const bf16_t* __restrict__ kr, const bf16_t* __restrict__ ki,
    const bf16_t* __restrict__ vt, bf16_t* __restrict__ ctx,
    int S, int D, int H, int DK) {
    __shared__ __align__(16) bf16_t pbuf[16 * 32];

    const int qt   = blockIdx.x;
    const int h    = blockIdx.y;
    const int b    = blockIdx.z;
    const int lane = threadIdx.x;
    const int half = (lane >> 4) & 1;
    const int koff = (lane & 16) ? 16 : 0;

    const float SCALE = 0.125f;   // 1/sqrt(DK)
    const float ALPHA = 0.3f;
    const float ISQDK = 0.125f;

    const size_t base  = ((size_t)b * S) * D + (size_t)h * DK;
    const size_t vbase = (((size_t)b * H + h) * DK) * S;
    const int q0 = qt * 16;

    v16bf aqr[2], aqi[2];
    {
        const bf16_t* qrp = qr + base + (size_t)(q0 + (lane & 15)) * D;
        const bf16_t* qip = qi + base + (size_t)(q0 + (lane & 15)) * D;
#pragma unroll
        for (int ks = 0; ks < 2; ++ks) {
            aqr[ks] = ldfragA(qrp + ks * 32, lane);
            aqi[ks] = ldfragA(qip + ks * 32, lane);
        }
    }

    v8f acc[4];
#pragma unroll
    for (int g = 0; g < 4; ++g) acc[g] = vzero8();
    float mrow[8], lrow[8];
#pragma unroll
    for (int r = 0; r < 8; ++r) { mrow[r] = -1e30f; lrow[r] = 0.0f; }

    for (int k0 = 0; k0 < S; k0 += 32) {
        if (k0 + 32 < S) {   // warm L0/L2 for next chunk
            __builtin_prefetch(kr + base + (size_t)(k0 + 32 + (lane & 31)) * D, 0, 1);
            __builtin_prefetch(ki + base + (size_t)(k0 + 32 + (lane & 31)) * D, 0, 1);
            __builtin_prefetch(vt + vbase + (size_t)(lane & 31) * S + k0 + 32, 0, 1);
        }
        float s[2][8];
#pragma unroll
        for (int t = 0; t < 2; ++t) {
            const int kb = k0 + t * 16;
            const bf16_t* krp = kr + base + (size_t)(kb + (lane & 15)) * D;
            const bf16_t* kip = ki + base + (size_t)(kb + (lane & 15)) * D;
            v8f sr = vzero8(), tp = vzero8(), tn = vzero8();
#pragma unroll
            for (int ks = 0; ks < 2; ++ks) {
                v16bf bkr = ldfragB(krp + ks * 32 + koff);
                v16bf bki = ldfragB(kip + ks * 32 + koff);
                sr = WMMA_BF16(aqr[ks], bkr, sr);
                sr = WMMA_BF16(aqi[ks], bki, sr);
                tp = WMMA_BF16(aqi[ks], bkr, tp);
                tn = WMMA_BF16(aqr[ks], bki, tn);
            }
#pragma unroll
            for (int r = 0; r < 8; ++r) {
                float srr = sr[r] * SCALE;
                float sii = (tp[r] - tn[r]) * SCALE;
                float mag = sqrtf(srr * srr + sii * sii);
                float ph  = (mag == 0.0f) ? 1.0f : (srr / mag);
                s[t][r] = (mag + ALPHA * ph) * ISQDK;
            }
        }
#pragma unroll
        for (int r = 0; r < 8; ++r) {
            float cm = fmaxf(s[0][r], s[1][r]);
#pragma unroll
            for (int m = 1; m < 16; m <<= 1) cm = fmaxf(cm, __shfl_xor(cm, m, 32));
            float mnew = fmaxf(mrow[r], cm);
            float sc   = __expf(mrow[r] - mnew);
            mrow[r] = mnew;
            float p0 = __expf(s[0][r] - mnew);
            float p1 = __expf(s[1][r] - mnew);
            float rs = p0 + p1;
#pragma unroll
            for (int m = 1; m < 16; m <<= 1) rs += __shfl_xor(rs, m, 32);
            lrow[r] = lrow[r] * sc + rs;
#pragma unroll
            for (int g = 0; g < 4; ++g) acc[g][r] *= sc;
            s[0][r] = p0; s[1][r] = p1;
        }
        __syncthreads();
#pragma unroll
        for (int r = 0; r < 8; ++r) {
            int row = r + half * 8;
            pbuf[row * 32 +      (lane & 15)] = (bf16_t)s[0][r];
            pbuf[row * 32 + 16 + (lane & 15)] = (bf16_t)s[1][r];
        }
        __syncthreads();
        v16bf pf = ldfragA(&pbuf[(lane & 15) * 32], lane);
#pragma unroll
        for (int g = 0; g < 4; ++g) {
            v16bf bv = ldfragB(vt + vbase + (size_t)(g * 16 + (lane & 15)) * S + k0 + koff);
            acc[g] = WMMA_BF16(pf, bv, acc[g]);
        }
        __syncthreads();
    }

#pragma unroll
    for (int g = 0; g < 4; ++g) {
#pragma unroll
        for (int r = 0; r < 8; ++r) {
            int row = q0 + r + half * 8;
            float o = acc[g][r] / lrow[r];
            ctx[base + (size_t)row * D + g * 16 + (lane & 15)] = (bf16_t)o;
        }
    }
}

// ---------------------------------------------------------------------------
// y = LayerNorm(x + h) * g + be  (row of D=512), f32 + bf16 outputs
// ---------------------------------------------------------------------------
__global__ __launch_bounds__(256) void add_ln_kernel(
    const float* __restrict__ x, const float* __restrict__ hres,
    const float* __restrict__ g, const float* __restrict__ be,
    float* __restrict__ y, bf16_t* __restrict__ yb, int D) {
    __shared__ float rsum[8], rsq[8];
    const int row  = blockIdx.x;
    const int tid  = threadIdx.x;
    const int wave = tid >> 5, lane = tid & 31;

    float v[2];
    float sum = 0.0f, sq = 0.0f;
#pragma unroll
    for (int i = 0; i < 2; ++i) {
        int d = tid + i * 256;
        float t = x[(size_t)row * D + d] + hres[(size_t)row * D + d];
        v[i] = t; sum += t; sq += t * t;
    }
#pragma unroll
    for (int m = 1; m < 32; m <<= 1) { sum += __shfl_xor(sum, m, 32); sq += __shfl_xor(sq, m, 32); }
    if (lane == 0) { rsum[wave] = sum; rsq[wave] = sq; }
    __syncthreads();
    if (tid == 0) {
        float a = 0.0f, c = 0.0f;
        for (int w = 0; w < 8; ++w) { a += rsum[w]; c += rsq[w]; }
        rsum[0] = a; rsq[0] = c;
    }
    __syncthreads();
    float mu   = rsum[0] / (float)D;
    float var  = rsq[0] / (float)D - mu * mu;
    float rstd = rsqrtf(var + 1e-5f);
#pragma unroll
    for (int i = 0; i < 2; ++i) {
        int d = tid + i * 256;
        float o = (v[i] - mu) * rstd * g[d] + be[d];
        y[(size_t)row * D + d] = o;
        yb[(size_t)row * D + d] = (bf16_t)o;
    }
}

// ---------------------------------------------------------------------------
// Head: out[row, c] = z[row,:] @ head_w[:,c] + head_b[c], C=2
// ---------------------------------------------------------------------------
__global__ __launch_bounds__(32) void head_kernel(
    const float* __restrict__ z, const float* __restrict__ hw,
    const float* __restrict__ hb, float* __restrict__ out, int D) {
    const int row = blockIdx.x, lane = threadIdx.x;
    float s0 = 0.0f, s1 = 0.0f;
    for (int d = lane; d < D; d += 32) {
        float zv = z[(size_t)row * D + d];
        s0 += zv * hw[d * 2 + 0];
        s1 += zv * hw[d * 2 + 1];
    }
#pragma unroll
    for (int m = 1; m < 32; m <<= 1) { s0 += __shfl_xor(s0, m, 32); s1 += __shfl_xor(s1, m, 32); }
    if (lane == 0) {
        out[row * 2 + 0] = s0 + hb[0];
        out[row * 2 + 1] = s1 + hb[1];
    }
}

// ---------------------------------------------------------------------------
// Host orchestration
// ---------------------------------------------------------------------------
extern "C" void kernel_launch(void* const* d_in, const int* in_sizes, int n_in,
                              void* d_out, int out_size, void* d_ws, size_t ws_size,
                              hipStream_t stream) {
    (void)in_sizes; (void)n_in; (void)out_size; (void)ws_size;
    const int Bb = 2, Ss = 2048, Dd = 512, Hh = 8, FF = 2048, Ll = 2, DK = 64;
    const size_t BS = (size_t)Bb * Ss;

    const int*   tokens = (const int*)d_in[0];
    const float* embed  = (const float*)d_in[1];
    const float* Wqr = (const float*)d_in[2];
    const float* Wqi = (const float*)d_in[3];
    const float* Wkr = (const float*)d_in[4];
    const float* Wki = (const float*)d_in[5];
    const float* Wv  = (const float*)d_in[6];
    const float* Wo  = (const float*)d_in[7];
    const float* bo  = (const float*)d_in[8];
    const float* W1  = (const float*)d_in[9];
    const float* b1  = (const float*)d_in[10];
    const float* W2  = (const float*)d_in[11];
    const float* b2  = (const float*)d_in[12];
    const float* g1  = (const float*)d_in[13];
    const float* be1 = (const float*)d_in[14];
    const float* g2  = (const float*)d_in[15];
    const float* be2 = (const float*)d_in[16];
    const float* hw  = (const float*)d_in[17];
    const float* hb  = (const float*)d_in[18];
    float* out = (float*)d_out;

    char* ws = (char*)d_ws;
    size_t off = 0;
    auto alloc = [&](size_t bytes) -> void* {
        void* p = ws + off;
        off += (bytes + 255) & ~(size_t)255;
        return p;
    };

    const size_t DD = (size_t)Dd * Dd;
    const size_t DF = (size_t)Dd * FF;
    bf16_t* wqr_t = (bf16_t*)alloc(Ll * DD * 2);
    bf16_t* wqi_t = (bf16_t*)alloc(Ll * DD * 2);
    bf16_t* wkr_t = (bf16_t*)alloc(Ll * DD * 2);
    bf16_t* wki_t = (bf16_t*)alloc(Ll * DD * 2);
    bf16_t* wv_t  = (bf16_t*)alloc(Ll * DD * 2);
    bf16_t* wo_t  = (bf16_t*)alloc(Ll * DD * 2);
    bf16_t* w1_t  = (bf16_t*)alloc(Ll * DF * 2);
    bf16_t* w2_t  = (bf16_t*)alloc(Ll * DF * 2);

    const size_t nAct = BS * Dd;
    float*  zA_f = (float*)alloc(nAct * 4);  bf16_t* zA_b = (bf16_t*)alloc(nAct * 2);
    float*  zB_f = (float*)alloc(nAct * 4);  bf16_t* zB_b = (bf16_t*)alloc(nAct * 2);
    bf16_t* zi_b = (bf16_t*)alloc(nAct * 2);
    float*  qr_f = (float*)alloc(nAct * 4);  bf16_t* qr_b = (bf16_t*)alloc(nAct * 2);
    float*  qi_f = (float*)alloc(nAct * 4);  bf16_t* qi_b = (bf16_t*)alloc(nAct * 2);
    float*  kr_f = (float*)alloc(nAct * 4);  bf16_t* kr_b = (bf16_t*)alloc(nAct * 2);
    float*  ki_f = (float*)alloc(nAct * 4);  bf16_t* ki_b = (bf16_t*)alloc(nAct * 2);
    float*  v_f  = (float*)alloc(nAct * 4);  bf16_t* v_b  = (bf16_t*)alloc(nAct * 2);
    bf16_t* vt_b  = (bf16_t*)alloc(nAct * 2);
    bf16_t* ctx_b = (bf16_t*)alloc(nAct * 2);
    float*  h_f   = (float*)alloc(nAct * 4);
    float*  z1_f  = (float*)alloc(nAct * 4); bf16_t* z1_b = (bf16_t*)alloc(nAct * 2);
    float*  fm_f  = (float*)alloc(BS * FF * 4);
    bf16_t* fm_b  = (bf16_t*)alloc(BS * FF * 2);

    auto cvtT = [&](const float* s, bf16_t* d, int K, int N) {
        size_t n = (size_t)K * N;
        cvt_transpose_kernel<<<dim3((unsigned)((n + 255) / 256)), dim3(256), 0, stream>>>(s, d, K, N);
    };
    for (int l = 0; l < Ll; ++l) {
        cvtT(Wqr + l * DD, wqr_t + l * DD, Dd, Dd);
        cvtT(Wqi + l * DD, wqi_t + l * DD, Dd, Dd);
        cvtT(Wkr + l * DD, wkr_t + l * DD, Dd, Dd);
        cvtT(Wki + l * DD, wki_t + l * DD, Dd, Dd);
        cvtT(Wv  + l * DD, wv_t  + l * DD, Dd, Dd);
        cvtT(Wo  + l * DD, wo_t  + l * DD, Dd, Dd);
        cvtT(W1  + l * DF, w1_t  + l * DF, Dd, FF);
        cvtT(W2  + l * DF, w2_t  + l * DF, FF, Dd);
    }

    embed_pos_kernel<<<dim3((unsigned)BS), dim3(256), 0, stream>>>(
        tokens, embed, zA_f, zA_b, zi_b, Ss, Dd);

    auto gemm = [&](const bf16_t* A, const bf16_t* Bt, const float* Cin, float* Cout,
                    bf16_t* Obf, const float* bias, int M, int N, int K,
                    float sign, int accum, int relu) {
        gemm_bf16_kernel<<<dim3(N / GN, M / GM), dim3(128), 0, stream>>>(
            A, Bt, Cin, Cout, Obf, bias, M, N, K, sign, accum, relu);
    };

    float*  zcur_f = zA_f; bf16_t* zcur_b = zA_b;
    float*  znxt_f = zB_f; bf16_t* znxt_b = zB_b;
    const int M = (int)BS;

    for (int l = 0; l < Ll; ++l) {
        const bf16_t* wqr_l = wqr_t + l * DD;
        const bf16_t* wqi_l = wqi_t + l * DD;
        const bf16_t* wkr_l = wkr_t + l * DD;
        const bf16_t* wki_l = wki_t + l * DD;
        const bf16_t* wv_l  = wv_t  + l * DD;
        const bf16_t* wo_l  = wo_t  + l * DD;
        const bf16_t* w1_l  = w1_t  + l * DF;
        const bf16_t* w2_l  = w2_t  + l * DF;

        if (l == 0) {
            gemm(zcur_b, wqr_l, nullptr, qr_f, nullptr, nullptr, M, Dd, Dd, +1.0f, 0, 0);
            gemm(zi_b,   wqi_l, qr_f,    qr_f, qr_b,    nullptr, M, Dd, Dd, -1.0f, 1, 0);
            gemm(zi_b,   wqr_l, nullptr, qi_f, nullptr, nullptr, M, Dd, Dd, +1.0f, 0, 0);
            gemm(zcur_b, wqi_l, qi_f,    qi_f, qi_b,    nullptr, M, Dd, Dd, +1.0f, 1, 0);
            gemm(zcur_b, wkr_l, nullptr, kr_f, nullptr, nullptr, M, Dd, Dd, +1.0f, 0, 0);
            gemm(zi_b,   wki_l, kr_f,    kr_f, kr_b,    nullptr, M, Dd, Dd, -1.0f, 1, 0);
            gemm(zi_b,   wkr_l, nullptr, ki_f, nullptr, nullptr, M, Dd, Dd, +1.0f, 0, 0);
            gemm(zcur_b, wki_l, ki_f,    ki_f, ki_b,    nullptr, M, Dd, Dd, +1.0f, 1, 0);
        } else {
            gemm(zcur_b, wqr_l, nullptr, qr_f, qr_b, nullptr, M, Dd, Dd, +1.0f, 0, 0);
            gemm(zcur_b, wqi_l, nullptr, qi_f, qi_b, nullptr, M, Dd, Dd, +1.0f, 0, 0);
            gemm(zcur_b, wkr_l, nullptr, kr_f, kr_b, nullptr, M, Dd, Dd, +1.0f, 0, 0);
            gemm(zcur_b, wki_l, nullptr, ki_f, ki_b, nullptr, M, Dd, Dd, +1.0f, 0, 0);
        }
        gemm(zcur_b, wv_l, nullptr, v_f, v_b, nullptr, M, Dd, Dd, +1.0f, 0, 0);
        transpose_v_kernel<<<dim3((unsigned)(nAct / 256)), dim3(256), 0, stream>>>(
            v_b, vt_b, Ss, Dd, Hh, DK);

        attn_kernel<<<dim3(Ss / 16, Hh, Bb), dim3(32), 0, stream>>>(
            qr_b, qi_b, kr_b, ki_b, vt_b, ctx_b, Ss, Dd, Hh, DK);

        gemm(ctx_b, wo_l, nullptr, h_f, nullptr, bo + (size_t)l * Dd, M, Dd, Dd, +1.0f, 0, 0);
        add_ln_kernel<<<dim3((unsigned)BS), dim3(256), 0, stream>>>(
            zcur_f, h_f, g1 + (size_t)l * Dd, be1 + (size_t)l * Dd, z1_f, z1_b, Dd);

        gemm(z1_b, w1_l, nullptr, fm_f, fm_b, b1 + (size_t)l * FF, M, FF, Dd, +1.0f, 0, 1);
        gemm(fm_b, w2_l, nullptr, h_f, nullptr, b2 + (size_t)l * Dd, M, Dd, FF, +1.0f, 0, 0);
        add_ln_kernel<<<dim3((unsigned)BS), dim3(256), 0, stream>>>(
            z1_f, h_f, g2 + (size_t)l * Dd, be2 + (size_t)l * Dd, znxt_f, znxt_b, Dd);

        float* tf = zcur_f; zcur_f = znxt_f; znxt_f = tf;
        bf16_t* tb = zcur_b; zcur_b = znxt_b; znxt_b = tb;
    }

    head_kernel<<<dim3((unsigned)BS), dim3(32), 0, stream>>>(zcur_f, hw, hb, out, Dd);
}
